// PixelRNN_66073776881698
// MI455X (gfx1250) — matrix operand order repaired and checked by reading, not compiled, not run on hardware
//
#include <hip/hip_runtime.h>
#include <math.h>

// PixelRNN on MI455X (gfx1250): diagonal BiLSTM with f16 WMMA (f32 accum).
// B=16, C=16, H=W=64, gates G=64, 7 layers, Ws=127 diagonal steps.
// Both diagonal directions run concurrently (grid = batch x dir) so the
// sequential scan chain is 7 passes, not 14.

#define Bn 16
#define Cn 16
#define Hn 64
#define Wn 64
#define Gn 64   // 4*C
#define NL 7
#define WSn 127 // H+W-1

typedef __attribute__((ext_vector_type(16))) _Float16 v16h;
typedef __attribute__((ext_vector_type(8)))  float    v8f;

// Hardware-native transcendentals (v_exp_f32 + v_rcp_f32): the scan is a
// latency-bound sequential loop, so short dependency chains matter more than
// ULP-exact libm here.
__device__ __forceinline__ float fsigm(float x) {
    return __builtin_amdgcn_rcpf(1.0f + __expf(-x));
}
__device__ __forceinline__ float ftanh(float x) {
    return 1.0f - 2.0f * __builtin_amdgcn_rcpf(__expf(2.0f * x) + 1.0f);
}

// ---------------------------------------------------------------------------
// Masked 7x7 conv, type A, 1 -> 16 channels. One thread per (b,h,w).
// ---------------------------------------------------------------------------
__global__ void k_conv_in(const float* __restrict__ X, const float* __restrict__ w_in,
                          const float* __restrict__ b_in, float* __restrict__ out) {
    int tid = blockIdx.x * blockDim.x + threadIdx.x;           // B*H*W = 65536
    int w = tid & 63, h = (tid >> 6) & 63, b = tid >> 12;
    float acc[Cn];
    #pragma unroll
    for (int co = 0; co < Cn; ++co) acc[co] = b_in[co];
    const float* xb = X + (size_t)b * Hn * Wn;
    for (int kh = 0; kh <= 3; ++kh) {
        int kwmax = (kh < 3) ? 7 : 3;                          // type A: exclude center
        int ih = h + kh - 3;
        if (ih < 0) continue;
        for (int kw = 0; kw < kwmax; ++kw) {
            int iw = w + kw - 3;
            if (iw < 0 || iw >= Wn) continue;
            float xv = xb[ih * Wn + iw];
            #pragma unroll
            for (int co = 0; co < Cn; ++co)
                acc[co] += w_in[co * 49 + kh * 7 + kw] * xv;
        }
    }
    #pragma unroll
    for (int co = 0; co < Cn; ++co)
        out[(((size_t)b * Cn + co) << 12) + (h << 6) + w] = acc[co];
}

// ---------------------------------------------------------------------------
// gin = Wis[64x16] x X[16 x (H*W)] + bias for BOTH directions, via WMMA f16.
// dir=1 reads x flipped along w. gin stored f16 in the dlstm (flipped) frame:
// [dir][B][64][H*W]. Block = 128 threads (4 waves), one 16x16 tile per wave.
// ---------------------------------------------------------------------------
__global__ void k_gin(const float* __restrict__ x, const float* __restrict__ wis_l,
                      const float* __restrict__ bis_l, _Float16* __restrict__ gin) {
    int wv   = threadIdx.x >> 5;
    int lane = threadIdx.x & 31;
    int row  = lane & 15;
    int half = lane >> 4;

    int t    = blockIdx.x * 4 + wv;    // 32768 tiles total (both directions)
    int dir  = t >> 14;
    int rem  = t & 16383;
    int b    = rem >> 10;
    int gt   = (rem >> 8) & 3;         // gate tile 0..3
    int pt   = rem & 255;              // position tile 0..255 (16 positions each)

    const float* wisp = wis_l + (size_t)dir * Gn * Cn;
    const float* bisp = bis_l + (size_t)dir * Gn;

    // A fragment: wis[g][c], K padded 16->32 with zeros (ISA 16-bit A 16x32 layout).
    int g = gt * 16 + row;
    v16h a;
    #pragma unroll
    for (int i = 0; i < 16; ++i) {
        int k = ((i < 8) ? i : i + 8) + half * 8;
        float v = (k < 16) ? wisp[g * 16 + k] : 0.0f;
        a[i] = (_Float16)v;
    }
    // B fragment: lane%16 = N (position); regs = K in same order as A. K>=16 is zero padding.
    v16h bf;
    {
        int pos = pt * 16 + row;
        int hh = pos >> 6, ww = pos & 63;
        int sw = dir ? (63 - ww) : ww;
        #pragma unroll
        for (int i = 0; i < 16; ++i) {
            int k = ((i < 8) ? i : i + 8) + half * 8;
            bf[i] = (k < 16) ? (_Float16)x[(((size_t)b * Cn + k) << 12) + (hh << 6) + sw]
                             : (_Float16)0.0f;
        }
    }
    v8f c = {};
    c = __builtin_amdgcn_wmma_f32_16x16x32_f16(false, a, false, bf, (short)0, c, false, false);

    _Float16* gout = gin + (((size_t)dir * Bn + b) * Gn << 12);
    int n = lane & 15;
    int pos = pt * 16 + n;
    #pragma unroll
    for (int r = 0; r < 8; ++r) {
        int gg = gt * 16 + r + 8 * half;
        gout[((size_t)gg << 12) + pos] = (_Float16)(c[r] + bisp[gg]);
    }
}

// ---------------------------------------------------------------------------
// Diagonal LSTM scan, both directions concurrently: block = (batch, dir),
// grid = 32. State lives in LDS. Per step: gates[64x64] =
// Wss[64x32] x [h_up;h][32x64] + gin  -> 16 WMMA tiles, one per wave
// (block = 512 threads = 16 waves). 127 sequential steps.
// ---------------------------------------------------------------------------
__global__ void k_scan(const _Float16* __restrict__ gin, const float* __restrict__ wss_l,
                       const float* __restrict__ bis_l, float* __restrict__ outL,
                       float* __restrict__ outR) {
    __shared__ float    s_hp[Cn][Hn];
    __shared__ float    s_cp[Cn][Hn];
    __shared__ _Float16 s_B[32][Hn];       // stacked [h_up(16); h(16)] x H
    __shared__ float    s_g[Gn][Hn];       // gates x H

    int b    = blockIdx.x & 15;
    int dir  = blockIdx.x >> 4;
    int tid  = threadIdx.x;                 // 0..511
    int wv   = tid >> 5;
    int lane = tid & 31;
    int row  = lane & 15;
    int half = lane >> 4;
    int gt   = wv >> 2;                     // gate tile 0..3
    int nt   = wv & 3;                      // H tile   0..3

    const float* wssp = wss_l + (size_t)dir * Gn * Cn * 2;
    const float* bisp = bis_l + (size_t)dir * Gn;
    const _Float16* ginb = gin + (((size_t)dir * Bn + b) * Gn << 12);
    float* out = dir ? outR : outL;

    // Preload A fragment (constant across all 127 steps):
    // A[g][k] = k<16 ? wss[g][k][0] (row-above tap) : wss[g][k-16][1] (same-row tap)
    int g = gt * 16 + row;
    v16h a;
    #pragma unroll
    for (int i = 0; i < 16; ++i) {
        int k = ((i < 8) ? i : i + 8) + half * 8;
        float v = (k < 16) ? wssp[(g * 16 + k) * 2 + 0]
                           : wssp[(g * 16 + (k - 16)) * 2 + 1];
        a[i] = (_Float16)v;
    }

    // zero state
    for (int idx = tid; idx < Cn * Hn; idx += 512) {
        s_hp[idx >> 6][idx & 63] = 0.0f;
        s_cp[idx >> 6][idx & 63] = 0.0f;
    }
    __syncthreads();

    for (int ws = 0; ws < WSn; ++ws) {
        // 1) build stacked K-matrix in LDS: rows 0..15 = h_up, rows 16..31 = h
        for (int idx = tid; idx < 32 * Hn; idx += 512) {
            int k = idx >> 6, n = idx & 63;
            float v = (k < 16) ? ((n > 0) ? s_hp[k][n - 1] : 0.0f)
                               : s_hp[k - 16][n];
            s_B[k][n] = (_Float16)v;
        }
        __syncthreads();

        // 2) per-wave WMMA tile
        v16h bf;
        #pragma unroll
        for (int n = 0; n < 16; ++n) bf[n] = s_B[((n < 8) ? n : n + 8) + half * 8][nt * 16 + row];

        int hcol = nt * 16 + row;
        int w = ws - hcol;
        bool valid = (w >= 0) && (w < Wn);
        v8f cf;
        #pragma unroll
        for (int r = 0; r < 8; ++r) {
            int gg = gt * 16 + r + 8 * half;
            cf[r] = valid ? (float)ginb[((size_t)gg << 12) + (hcol << 6) + w]
                          : bisp[gg];
        }
        // Hide L2 latency of next diagonal's gin column behind the WMMA.
        int wn = ws + 1 - hcol;
        if (wn >= 0 && wn < Wn)
            __builtin_prefetch(&ginb[((size_t)(gt * 16 + 8 * half) << 12) + (hcol << 6) + wn],
                               0, 1);

        cf = __builtin_amdgcn_wmma_f32_16x16x32_f16(false, a, false, bf, (short)0, cf,
                                                    false, false);
        #pragma unroll
        for (int r = 0; r < 8; ++r)
            s_g[gt * 16 + r + 8 * half][hcol] = cf[r];
        __syncthreads();

        // 3) gates -> state update -> scatter valid outputs (unskew)
        for (int idx = tid; idx < Cn * Hn; idx += 512) {
            int cc = idx >> 6, hh = idx & 63;
            float o  = s_g[cc][hh];
            float f  = s_g[16 + cc][hh];
            float ii = s_g[32 + cc][hh];
            float gg = s_g[48 + cc][hh];
            float cn = fsigm(f) * s_cp[cc][hh] + fsigm(ii) * ftanh(gg);
            float hn = fsigm(o) * ftanh(cn);
            s_cp[cc][hh] = cn;
            s_hp[cc][hh] = hn;
            int wv2 = ws - hh;
            if (wv2 >= 0 && wv2 < Wn) {
                int wout = dir ? (63 - wv2) : wv2;   // un-flip for right direction
                out[(((size_t)b * Cn + cc) << 12) + (hh << 6) + wout] = hn;
            }
        }
        __syncthreads();
    }
}

// ---------------------------------------------------------------------------
// h = x + left + shift_down_1_row(right)   (residual BiLSTM combine, in-place on h)
// ---------------------------------------------------------------------------
__global__ void k_combine(float* __restrict__ hbuf, const float* __restrict__ L,
                          const float* __restrict__ R) {
    int tid = blockIdx.x * blockDim.x + threadIdx.x;   // B*C*H*W
    int w = tid & 63, h = (tid >> 6) & 63;
    int bc = tid >> 12;
    float r = (h > 0) ? R[((size_t)bc << 12) + ((h - 1) << 6) + w] : 0.0f;
    hbuf[tid] = hbuf[tid] + L[tid] + r;
}

// ---------------------------------------------------------------------------
// Masked 7x7 conv, type B, 16 -> 16 channels, + ReLU. One thread per (b,co,h,w).
// ---------------------------------------------------------------------------
__global__ void k_midconv(const float* __restrict__ in, const float* __restrict__ wm,
                          const float* __restrict__ bm, float* __restrict__ out) {
    int tid = blockIdx.x * blockDim.x + threadIdx.x;   // B*C*H*W
    int w = tid & 63, h = (tid >> 6) & 63, co = (tid >> 12) & 15, b = tid >> 16;
    float acc = bm[co];
    for (int ci = 0; ci < Cn; ++ci) {
        const float* xb = in + (((size_t)b * Cn + ci) << 12);
        const float* wp = wm + (co * Cn + ci) * 49;
        for (int kh = 0; kh <= 3; ++kh) {
            int kwmax = (kh < 3) ? 7 : 4;              // type B: include center
            int ih = h + kh - 3;
            if (ih < 0) continue;
            for (int kw = 0; kw < kwmax; ++kw) {
                int iw = w + kw - 3;
                if (iw < 0 || iw >= Wn) continue;
                acc += wp[kh * 7 + kw] * xb[(ih << 6) + iw];
            }
        }
    }
    out[tid] = fmaxf(acc, 0.0f);
}

// ---------------------------------------------------------------------------
// 1x1 output conv: 16 -> 1 channel.
// ---------------------------------------------------------------------------
__global__ void k_outconv(const float* __restrict__ in, const float* __restrict__ wo,
                          const float* __restrict__ bo, float* __restrict__ out) {
    int tid = blockIdx.x * blockDim.x + threadIdx.x;   // B*H*W
    int hw = tid & 4095, b = tid >> 12;
    float acc = bo[0];
    #pragma unroll
    for (int c = 0; c < Cn; ++c)
        acc += wo[c] * in[(((size_t)b * Cn + c) << 12) + hw];
    out[tid] = acc;
}

// ---------------------------------------------------------------------------
extern "C" void kernel_launch(void* const* d_in, const int* in_sizes, int n_in,
                              void* d_out, int out_size, void* d_ws, size_t ws_size,
                              hipStream_t stream) {
    const float* X     = (const float*)d_in[0];
    const float* w_in  = (const float*)d_in[1];
    const float* b_in  = (const float*)d_in[2];
    const float* wis   = (const float*)d_in[3];
    const float* bis   = (const float*)d_in[4];
    const float* wss   = (const float*)d_in[5];
    const float* w_mid = (const float*)d_in[6];
    const float* b_mid = (const float*)d_in[7];
    const float* w_out = (const float*)d_in[8];
    const float* b_out = (const float*)d_in[9];
    float* out = (float*)d_out;

    const size_t NH = (size_t)Bn * Cn * Hn * Wn;     // 1,048,576 floats
    float* hbuf = (float*)d_ws;
    float* lbuf = hbuf + NH;
    float* rbuf = lbuf + NH;
    _Float16* ginb = (_Float16*)(rbuf + NH);          // 2 dirs * B*64*H*W f16 = 16 MB

    k_conv_in<<<256, 256, 0, stream>>>(X, w_in, b_in, hbuf);

    for (int l = 0; l < NL; ++l) {
        const float* wis_l = wis + (size_t)l * 2 * Gn * Cn;
        const float* bis_l = bis + (size_t)l * 2 * Gn;
        const float* wss_l = wss + (size_t)l * 2 * Gn * Cn * 2;
        k_gin<<<8192, 128, 0, stream>>>(hbuf, wis_l, bis_l, ginb);
        k_scan<<<32, 512, 0, stream>>>(ginb, wss_l, bis_l, lbuf, rbuf);
        k_combine<<<4096, 256, 0, stream>>>(hbuf, lbuf, rbuf);
    }

    k_midconv<<<4096, 256, 0, stream>>>(hbuf, w_mid, b_mid, lbuf);
    k_midconv<<<4096, 256, 0, stream>>>(lbuf, w_mid + 16 * 16 * 49, b_mid + 16, rbuf);
    k_outconv<<<256, 256, 0, stream>>>(rbuf, w_out, b_out, out);
}
